// ScaledDotProductAttentionFunction_77000173683359
// MI455X (gfx1250) — matrix operand order repaired and checked by reading, not compiled, not run on hardware
//
#include <hip/hip_runtime.h>

typedef _Float16 v16h __attribute__((ext_vector_type(16)));
typedef _Float16 v8h  __attribute__((ext_vector_type(8)));
typedef _Float16 v4h  __attribute__((ext_vector_type(4)));
typedef _Float16 v2h  __attribute__((ext_vector_type(2)));
typedef float    v8f  __attribute__((ext_vector_type(8)));
typedef float    v4f  __attribute__((ext_vector_type(4)));

#define NUM_HEADS   8
#define HEAD_DIM    64
#define TOTAL_TOK   49152
#define M_TILE      128     // query rows per workgroup (8 waves x 16 rows)
#define KC          64      // key chunk per iteration
#define WAVES       8

// 1/sqrt(64) * log2(e): QK^T then directly exp2-able scores
#define SCALE_LOG2E 0.18033688f

// raw v_exp_f32 (exp2)
__device__ __forceinline__ float exp2_fast(float x) {
    return __builtin_amdgcn_exp2f(x);
}

// xor-reduce max within 16-lane groups via ds_swizzle (group-of-32 xor mode).
__device__ __forceinline__ float swz_max16(float x) {
    x = fmaxf(x, __int_as_float(__builtin_amdgcn_ds_swizzle(__float_as_int(x), 0x041F)));
    x = fmaxf(x, __int_as_float(__builtin_amdgcn_ds_swizzle(__float_as_int(x), 0x081F)));
    x = fmaxf(x, __int_as_float(__builtin_amdgcn_ds_swizzle(__float_as_int(x), 0x101F)));
    x = fmaxf(x, __int_as_float(__builtin_amdgcn_ds_swizzle(__float_as_int(x), 0x201F)));
    return x;
}

// pack 8 f32 -> 8 f16 with v_cvt_pk (RTZ)
__device__ __forceinline__ v8h pk8(v4f a, v4f b) {
    auto p0 = __builtin_amdgcn_cvt_pkrtz(a[0], a[1]);
    auto p1 = __builtin_amdgcn_cvt_pkrtz(a[2], a[3]);
    auto p2 = __builtin_amdgcn_cvt_pkrtz(b[0], b[1]);
    auto p3 = __builtin_amdgcn_cvt_pkrtz(b[2], b[3]);
    v8h r;
    r[0] = (_Float16)p0[0]; r[1] = (_Float16)p0[1];
    r[2] = (_Float16)p1[0]; r[3] = (_Float16)p1[1];
    r[4] = (_Float16)p2[0]; r[5] = (_Float16)p2[1];
    r[6] = (_Float16)p3[0]; r[7] = (_Float16)p3[1];
    return r;
}

#define WMMA_F16(A, B, C) \
    __builtin_amdgcn_wmma_f32_16x16x32_f16(false, (A), false, (B), (short)0, (C), false, false)

__global__ __launch_bounds__(256)
void sdpa_flash_wmma_kernel(const float* __restrict__ Q,
                            const float* __restrict__ K,
                            const float* __restrict__ V,
                            float* __restrict__ O)
{
    __shared__ __align__(128) _Float16 Qs[M_TILE][HEAD_DIM];   // 16 KB
    __shared__ __align__(128) _Float16 Ks[KC][HEAD_DIM];       //  8 KB (keys row-major)
    __shared__ __align__(128) _Float16 Vt[HEAD_DIM][KC];       //  8 KB (V^T, key-slot permuted)
    __shared__ __align__(128) _Float16 Ps[WAVES][16][KC];      // 16 KB (P, key-slot permuted)

    const int tid  = threadIdx.x;
    const int wave = tid >> 5;
    const int lane = tid & 31;
    const int lh   = lane >> 4;
    const int ln   = lane & 15;

    // ---- decode block -> (segment, batch, head, m-tile); 1024 blocks/segment ----
    const int blk = blockIdx.x;
    const int seg = blk >> 10;
    const int idx = blk & 1023;
    const int S      = 512 << seg;
    const int mtiles = 4 << seg;
    const int b   = idx / (NUM_HEADS * mtiles);
    const int rem = idx - b * (NUM_HEADS * mtiles);
    const int h   = rem / mtiles;
    const int mt  = rem - h * mtiles;
    const int seg_off = seg << 14;

    const long headbase = (long)h * TOTAL_TOK;
    const float* Qh = Q + headbase * HEAD_DIM;
    const float* Kh = K + headbase * HEAD_DIM;
    const float* Vh = V + headbase * HEAD_DIM;
    float*       Oh = O + headbase * HEAD_DIM;

    const long t0  = (long)seg_off + (long)b * S + (long)mt * M_TILE;
    const long kv0 = (long)seg_off + (long)b * S;

    // ---- stage Q tile (scaled by 1/8 * log2e) as f16: 1 row x 32 cols/thread ----
    {
        const int r  = tid >> 1;
        const int cq = (tid & 1) * 32;
        const float* src = Qh + (t0 + r) * HEAD_DIM + cq;
        #pragma unroll
        for (int j = 0; j < 4; ++j) {
            v4f a = *(const v4f*)(src + j * 8);
            v4f c = *(const v4f*)(src + j * 8 + 4);
            *(v8h*)&Qs[r][cq + j * 8] = pk8(a * SCALE_LOG2E, c * SCALE_LOG2E);
        }
    }

    // all-ones B operand for row-sum WMMA
    v16h ones;
    #pragma unroll
    for (int j = 0; j < 16; ++j) ones[j] = (_Float16)1.0f;

    float m_row[8];
    v8f lrow = {};
    #pragma unroll
    for (int i = 0; i < 8; ++i) m_row[i] = -3.0e38f;
    v8f o0 = {}, o1 = {}, o2 = {}, o3 = {};

    const int qrow = (wave << 4) | ln;
    const int kb1  = lh * 16;           // A-operand K bases: {0,8} / {16,24}

    const int nchunks = S / KC;
    for (int kc = 0; kc < nchunks; ++kc) {
        __syncthreads();
        // ---- stage K (row-major) : 1 key row x 16 cols per thread ----
        {
            const int rk = tid >> 2;
            const int ck = (tid & 3) * 16;
            const float* ksrc = Kh + (kv0 + (long)(kc * KC + rk)) * HEAD_DIM + ck;
            #pragma unroll
            for (int j = 0; j < 2; ++j) {
                v4f a = *(const v4f*)(ksrc + j * 8);
                v4f c = *(const v4f*)(ksrc + j * 8 + 4);
                *(v8h*)&Ks[rk][ck + j * 8] = pk8(a, c);
            }
            __builtin_prefetch(ksrc + KC * HEAD_DIM, 0, 3);
        }
        // ---- stage V^T with key-slot permutation: slot(k) = 4*(k&15) + (k>>4) ----
        {
            const int rr = tid >> 3;
            const int rv = (rr & 15) + 32 * (rr >> 4);   // rows rv, rv+16 -> adjacent slots
            const int cv = (tid & 7) * 8;
            const int js = 4 * (rv & 15) + (rv >> 4);    // even
            const float* v0p = Vh + (kv0 + (long)(kc * KC + rv)) * HEAD_DIM + cv;
            const float* v1p = v0p + 16 * HEAD_DIM;
            v4f a0 = *(const v4f*)v0p, a1 = *(const v4f*)(v0p + 4);
            v4f c0 = *(const v4f*)v1p, c1 = *(const v4f*)(v1p + 4);
            #pragma unroll
            for (int j = 0; j < 4; ++j) {
                auto p = __builtin_amdgcn_cvt_pkrtz(a0[j], c0[j]);
                v2h q; q[0] = (_Float16)p[0]; q[1] = (_Float16)p[1];
                *(v2h*)&Vt[cv + j][js] = q;
            }
            #pragma unroll
            for (int j = 0; j < 4; ++j) {
                auto p = __builtin_amdgcn_cvt_pkrtz(a1[j], c1[j]);
                v2h q; q[0] = (_Float16)p[0]; q[1] = (_Float16)p[1];
                *(v2h*)&Vt[cv + 4 + j][js] = q;
            }
            __builtin_prefetch(v0p + KC * HEAD_DIM, 0, 3);
        }
        __syncthreads();

        // ---- S = Q K^T : 16 rows x 64 keys (4 N-tiles), contract D=64 in 2 chunks ----
        v8f c0 = {}, c1 = {}, c2 = {}, c3 = {};
        #pragma unroll
        for (int dc = 0; dc < 2; ++dc) {
            v16h aq;
            {
                const v8h lo = *(const v8h*)&Qs[qrow][dc * 32 + kb1];
                const v8h hi = *(const v8h*)&Qs[qrow][dc * 32 + kb1 + 8];
                #pragma unroll
                for (int j = 0; j < 8; ++j) { aq[j] = lo[j]; aq[8 + j] = hi[j]; }
            }
            const int ko = dc * 32 + lh * 16;
            const v16h bk0 = *(const v16h*)&Ks[ln][ko];
            const v16h bk1 = *(const v16h*)&Ks[16 + ln][ko];
            const v16h bk2 = *(const v16h*)&Ks[32 + ln][ko];
            const v16h bk3 = *(const v16h*)&Ks[48 + ln][ko];
            c0 = WMMA_F16(aq, bk0, c0);
            c1 = WMMA_F16(aq, bk1, c1);
            c2 = WMMA_F16(aq, bk2, c2);
            c3 = WMMA_F16(aq, bk3, c3);
        }

        // ---- online softmax (log2 domain): row max via ds_swizzle ----
        float mnew[8];
        bool grow = false;
        #pragma unroll
        for (int i = 0; i < 8; ++i) {
            float mx = fmaxf(fmaxf(c0[i], c1[i]), fmaxf(c2[i], c3[i]));
            mx = swz_max16(mx);
            grow = grow || (mx > m_row[i]);
            mnew[i] = fmaxf(m_row[i], mx);
        }
        // rescale only when some row's max actually grew (wave-uniform branch)
        if (__builtin_amdgcn_ballot_w32(grow)) {
            v8f alpha;
            #pragma unroll
            for (int i = 0; i < 8; ++i) alpha[i] = exp2_fast(m_row[i] - mnew[i]);
            o0 *= alpha; o1 *= alpha; o2 *= alpha; o3 *= alpha;
            lrow *= alpha;
        }
        #pragma unroll
        for (int i = 0; i < 8; ++i) {
            m_row[i] = mnew[i];
            const float p0 = exp2_fast(c0[i] - mnew[i]);
            const float p1 = exp2_fast(c1[i] - mnew[i]);
            const float p2 = exp2_fast(c2[i] - mnew[i]);
            const float p3 = exp2_fast(c3[i] - mnew[i]);
            auto plo = __builtin_amdgcn_cvt_pkrtz(p0, p1);
            auto phi = __builtin_amdgcn_cvt_pkrtz(p2, p3);
            v4h pv;
            pv[0] = (_Float16)plo[0]; pv[1] = (_Float16)plo[1];
            pv[2] = (_Float16)phi[0]; pv[3] = (_Float16)phi[1];
            *(v4h*)&Ps[wave][i + 8 * lh][4 * ln] = pv;   // one b64 store
        }

        // ---- O += P V ; row-sum via P x ones ----
        v16h ap[2];
        #pragma unroll
        for (int kk = 0; kk < 2; ++kk) {
            const v8h lo = *(const v8h*)&Ps[wave][ln][kk * 32 + kb1];
            const v8h hi = *(const v8h*)&Ps[wave][ln][kk * 32 + kb1 + 8];
            #pragma unroll
            for (int j = 0; j < 8; ++j) { ap[kk][j] = lo[j]; ap[kk][8 + j] = hi[j]; }
        }
        v8f rs = {};
        #pragma unroll
        for (int kk = 0; kk < 2; ++kk) {
            const int ko = kk * 32 + lh * 16;
            const v16h bv0 = *(const v16h*)&Vt[ln][ko];
            const v16h bv1 = *(const v16h*)&Vt[16 + ln][ko];
            const v16h bv2 = *(const v16h*)&Vt[32 + ln][ko];
            const v16h bv3 = *(const v16h*)&Vt[48 + ln][ko];
            o0 = WMMA_F16(ap[kk], bv0, o0);
            o1 = WMMA_F16(ap[kk], bv1, o1);
            o2 = WMMA_F16(ap[kk], bv2, o2);
            o3 = WMMA_F16(ap[kk], bv3, o3);
            rs = WMMA_F16(ap[kk], ones, rs);   // per-row sum, broadcast across lanes
        }
        lrow += rs;
    }

    // ---- normalize and write O (fp32) ----
    #pragma unroll
    for (int i = 0; i < 8; ++i) {
        const float inv = 1.0f / lrow[i];
        const long tok = t0 + (wave << 4) + i + 8 * lh;
        float* dst = Oh + tok * HEAD_DIM + ln;
        dst[0]  = o0[i] * inv;
        dst[16] = o1[i] * inv;
        dst[32] = o2[i] * inv;
        dst[48] = o3[i] * inv;
    }
}

extern "C" void kernel_launch(void* const* d_in, const int* in_sizes, int n_in,
                              void* d_out, int out_size, void* d_ws, size_t ws_size,
                              hipStream_t stream) {
    const float* Q = (const float*)d_in[0];
    const float* K = (const float*)d_in[1];
    const float* V = (const float*)d_in[2];
    float*       O = (float*)d_out;
    dim3 grid(3072), block(256);
    hipLaunchKernelGGL(sdpa_flash_wmma_kernel, grid, block, 0, stream, Q, K, V, O);
}
